// SelfAttention_34926674051290
// MI455X (gfx1250) — matrix operand order repaired and checked by reading, not compiled, not run on hardware
//
#include <hip/hip_runtime.h>
#include <hip/hip_bf16.h>

// ---------------------------------------------------------------------------
// Self-attention (B=4, S=2048, H=8, Dh=256) for gfx1250 / MI455X.
// bf16 WMMA (v_wmma_f32_16x16x32_bf16), f32 accumulate.
// Flash attention: K block staged by the Tensor Data Mover (TENSORcnt),
// V block by per-lane GLOBAL_LOAD_ASYNC_TO_LDS (ASYNCcnt), double-buffered.
// ---------------------------------------------------------------------------

#define BATCH   4
#define SEQ     2048
#define DHEAD   256
#define NHEAD   8
#define MROWS   (BATCH * SEQ)          // 8192
#define NQKV    (3 * NHEAD * DHEAD)    // 6144
#define HD      (NHEAD * DHEAD)        // 2048
#define VT_STRIDE 40                   // halfwords; keeps 16B alignment, skews banks

#define AS1 __attribute__((address_space(1)))
#define AS3 __attribute__((address_space(3)))

typedef __attribute__((ext_vector_type(16))) __bf16        v16bf;
typedef __attribute__((ext_vector_type(8)))  float         v8f;
typedef __attribute__((ext_vector_type(4)))  int           v4i_t;
typedef __attribute__((ext_vector_type(8)))  int           v8i_t;
typedef __attribute__((ext_vector_type(4)))  unsigned int  v4u_t;

union FragBF {
    v16bf v;
    uint4 q[2];
    unsigned short h[16];
};
union U4H {
    uint4 q;
    unsigned short h[8];
};

static __device__ inline unsigned short f32_to_bf16(float f) {
    union { float f; unsigned u; } v; v.f = f;
    unsigned r = v.u + 0x7FFFu + ((v.u >> 16) & 1u);   // round-to-nearest-even
    return (unsigned short)(r >> 16);
}
static __device__ inline float bf16_to_f32(unsigned short h) {
    union { unsigned u; float f; } v; v.u = ((unsigned)h) << 16;
    return v.f;
}

static __device__ inline v8f wmma_bf16(v16bf a, v16bf b, v8f c) {
    // (neg_a, A, neg_b, B, c_mod, C, reuse_a, reuse_b)
    return __builtin_amdgcn_wmma_f32_16x16x32_bf16(false, a, false, b,
                                                   (short)0, c, false, false);
}

// ---- Async global->LDS copy (16 bytes per lane), ASYNCcnt-tracked ---------
#if __has_builtin(__builtin_amdgcn_global_load_async_to_lds_b128)
#define HAVE_ASYNC_LDS 1
static __device__ inline void async_copy16(const unsigned short* g, unsigned short* l) {
    __builtin_amdgcn_global_load_async_to_lds_b128(
        (AS1 v4i_t*)g, (AS3 v4i_t*)l, 0, 0);
}
#else
#define HAVE_ASYNC_LDS 0
static __device__ inline void async_copy16(const unsigned short* g, unsigned short* l) {
    *(uint4*)l = *(const uint4*)g;                     // sync fallback
}
#endif

#if __has_builtin(__builtin_amdgcn_s_wait_asynccnt)
#define WAIT_ASYNC(n) __builtin_amdgcn_s_wait_asynccnt(n)
#else
#define WAIT_ASYNC(n) asm volatile("s_wait_asynccnt %0" :: "i"(n) : "memory")
#endif

// ---- Tensor Data Mover: 2D tile (32 rows x 256 cols, bf16) to LDS ---------
#if __has_builtin(__builtin_amdgcn_tensor_load_to_lds)
#define HAVE_TDM 1
static __device__ inline void tdm_load_tile32(const unsigned short* gsrc,
                                              unsigned short* ldst) {
    const unsigned long long ga = (unsigned long long)(uintptr_t)gsrc;
    const unsigned lds = (unsigned)(uintptr_t)ldst;    // generic low 32b = LDS addr
    v4u_t g0;
    g0[0] = 1u;                                        // count=1, user mode
    g0[1] = lds;                                       // lds_addr
    g0[2] = (unsigned)(ga & 0xFFFFFFFFu);              // global_addr[31:0]
    g0[3] = (unsigned)((ga >> 32) & 0x1FFFFFFu)        // global_addr[56:32]
          | (2u << 30);                                // type = 2 ("image")
    v8i_t g1;
    g1[0] = 0x00010000;          // wg_mask=0, data_size=1 (2 bytes), no flags
    g1[1] = (int)(256u << 16);   // tensor_dim0 = 256   (bits 79:48, low half)
    g1[2] = (int)(2048u << 16);  // dim0 hi = 0; tensor_dim1 = 2048 (bits 111:80)
    g1[3] = (int)(256u << 16);   // dim1 hi = 0; tile_dim0 = 256  (bits 127:112)
    g1[4] = 32;                  // tile_dim1 = 32; tile_dim2 = 0
    g1[5] = 256;                 // tensor_dim0_stride = 256 (bits 207:160, low)
    g1[6] = 0;                   // stride hi, tensor_dim1_stride lo
    g1[7] = 0;
    // clang-23 6-arg form: (g0, g1, g2, g3, extra, cpol); unused groups zero.
    __builtin_amdgcn_tensor_load_to_lds(g0, g1, (v4i_t)0, (v4i_t)0, (v8i_t)0, 0);
}
#else
#define HAVE_TDM 0
#endif

static __device__ inline void wait_tensor(void) {
#if __has_builtin(__builtin_amdgcn_s_wait_tensorcnt)
    __builtin_amdgcn_s_wait_tensorcnt(1);   // <=1 outstanding: previous TDM done
#else
    asm volatile("s_wait_tensorcnt 1" ::: "memory");
#endif
}
static __device__ inline void wait_tensor0(void) {
#if __has_builtin(__builtin_amdgcn_s_wait_tensorcnt)
    __builtin_amdgcn_s_wait_tensorcnt(0);
#else
    asm volatile("s_wait_tensorcnt 0" ::: "memory");
#endif
}

// ---------------------------------------------------------------------------
// Kernel 1: f32 -> bf16 conversion (grid-stride)
// ---------------------------------------------------------------------------
__global__ void cvt_bf16_kernel(const float* __restrict__ src,
                                unsigned short* __restrict__ dst, int n) {
    int i = blockIdx.x * blockDim.x + threadIdx.x;
    int stride = gridDim.x * blockDim.x;
    for (; i < n; i += stride) dst[i] = f32_to_bf16(src[i]);
}

// ---------------------------------------------------------------------------
// Kernel 2: QKV projection.  C[m,n] = sum_k Xb[m,k]*Wqkv[n,k] + b_qkv[n]
// ---------------------------------------------------------------------------
__global__ __launch_bounds__(128) void gemm_qkv_kernel(
        const unsigned short* __restrict__ Xb,
        const unsigned short* __restrict__ Wb,
        const float* __restrict__ bias,
        unsigned short* __restrict__ qkv) {
    const int tid  = threadIdx.x;
    const int wave = tid >> 5;
    const int lane = tid & 31;
    const int lm   = lane & 15;
    const int half = lane >> 4;

    const int m0 = blockIdx.y * 64 + wave * 16;
    const int n0 = blockIdx.x * 64;

    v8f acc[4];
#pragma unroll
    for (int j = 0; j < 4; ++j)
#pragma unroll
        for (int i = 0; i < 8; ++i) acc[j][i] = 0.0f;

    const unsigned short* arow = Xb + (size_t)(m0 + lm) * DHEAD;

    FragBF a[2], bmf[2][4];
    auto loadFrags = [&](int p, int kc) {
        a[p].q[0] = *(const uint4*)(arow + kc + 16 * half);
        a[p].q[1] = *(const uint4*)(arow + kc + 16 * half + 8);
#pragma unroll
        for (int j = 0; j < 4; ++j) {
            const unsigned short* wr =
                Wb + (size_t)(n0 + j * 16 + lm) * DHEAD + kc + 16 * half;
            bmf[p][j].q[0] = *(const uint4*)(wr);
            bmf[p][j].q[1] = *(const uint4*)(wr + 8);
        }
    };

    loadFrags(0, 0);
    int p = 0;
    for (int kc = 0; kc < DHEAD; kc += 32, p ^= 1) {
        if (kc + 32 < DHEAD) loadFrags(p ^ 1, kc + 32);   // overlap with WMMAs
#pragma unroll
        for (int j = 0; j < 4; ++j)
            acc[j] = wmma_bf16(a[p].v, bmf[p][j].v, acc[j]);
    }

#pragma unroll
    for (int j = 0; j < 4; ++j) {
        const int n  = n0 + j * 16 + lm;
        const float bv = bias[n];
        const int t = n >> 11;            // n / 2048  -> q/k/v
        const int h = (n >> 8) & 7;       // head
        const int d = n & 255;            // dim in head
#pragma unroll
        for (int r = 0; r < 8; ++r) {
            const int row = m0 + r + 8 * half;
            const int b = row >> 11;      // row / 2048
            const int s = row & 2047;
            size_t addr = (((size_t)t * BATCH + b) * NHEAD + h) * (size_t)(SEQ * DHEAD)
                        + (size_t)s * DHEAD + d;
            qkv[addr] = f32_to_bf16(acc[j][r] + bv);
        }
    }
}

// ---------------------------------------------------------------------------
// Kernel 3: RoPE in-place on Q and K (t = 0,1).  Pair (i, i+128), dim=256.
// ---------------------------------------------------------------------------
__global__ void rope_kernel(unsigned short* __restrict__ qkv) {
    const size_t total = (size_t)2 * BATCH * NHEAD * SEQ * (DHEAD / 2); // 16.78M
    size_t idx = (size_t)blockIdx.x * blockDim.x + threadIdx.x;
    const size_t stride = (size_t)gridDim.x * blockDim.x;
    for (; idx < total; idx += stride) {
        const int i = (int)(idx & 127);
        size_t rest = idx >> 7;
        const int s = (int)(rest & 2047); rest >>= 11;
        const int h = (int)(rest & 7);    rest >>= 3;
        const int b = (int)(rest & 3);    rest >>= 2;
        const int t = (int)rest;          // 0=Q, 1=K
        unsigned short* p = qkv
            + (((size_t)t * BATCH + b) * NHEAD + h) * (size_t)(SEQ * DHEAD)
            + (size_t)s * DHEAD;
        const float invf = __powf(10000.0f, -(2.0f * (float)i) / 256.0f);
        float sn, cs;
        __sincosf((float)s * invf, &sn, &cs);
        const float x1 = bf16_to_f32(p[i]);
        const float x2 = bf16_to_f32(p[i + 128]);
        p[i]       = f32_to_bf16(x1 * cs - x2 * sn);
        p[i + 128] = f32_to_bf16(x1 * sn + x2 * cs);
    }
}

// ---------------------------------------------------------------------------
// Kernel 4: Flash attention.  K staged by TDM, V by per-lane async copies,
// double-buffered LDS.  One workgroup = 4 waves, 16 query rows per wave.
// ---------------------------------------------------------------------------
__global__ __launch_bounds__(128) void flash_attn_kernel(
        const unsigned short* __restrict__ qkv,
        unsigned short* __restrict__ obuf) {
    __shared__ unsigned short lds_k[2][32 * DHEAD];       // 2 x 16 KB, row-major
    __shared__ unsigned short lds_v[2][32 * DHEAD];       // 2 x 16 KB, row-major
    __shared__ unsigned short lds_vt[DHEAD * VT_STRIDE];  // 20 KB, transposed V
    __shared__ unsigned short lds_p[4 * 16 * 32];         // 4 KB, per-wave P tiles

    const int tid  = threadIdx.x;
    const int wave = tid >> 5;
    const int lane = tid & 31;
    const int lm   = lane & 15;
    const int half = lane >> 4;

    const int h = blockIdx.y;
    const int b = blockIdx.z;
    const int q0 = blockIdx.x * 64 + wave * 16;

    const size_t headStride = (size_t)SEQ * DHEAD;
    const unsigned short* Qg = qkv + (((size_t)0 * BATCH + b) * NHEAD + h) * headStride;
    const unsigned short* Kg = qkv + (((size_t)1 * BATCH + b) * NHEAD + h) * headStride;
    const unsigned short* Vg = qkv + (((size_t)2 * BATCH + b) * NHEAD + h) * headStride;

    // V block: 8 async B128 issues per thread (ASYNCcnt).
    auto stage_v = [&](const unsigned short* vg, unsigned short* ldsV) {
#pragma unroll
        for (int j = 0; j < 8; ++j) {
            const int e0 = (tid + j * 128) * 8;
            async_copy16(vg + e0, ldsV + e0);
        }
    };
    // K block: one TDM descriptor from wave 0 (TENSORcnt), or async fallback.
    auto stage_k = [&](const unsigned short* kg, unsigned short* ldsK) {
#if HAVE_TDM
        if (wave == 0) tdm_load_tile32(kg, ldsK);
#else
#pragma unroll
        for (int j = 0; j < 8; ++j) {
            const int e0 = (tid + j * 128) * 8;
            async_copy16(kg + e0, ldsK + e0);
        }
#endif
    };

    // ---- Load Q fragments (A layout), prescaled by 1/sqrt(256) = 1/16 ----
    FragBF qf[8];
    {
        const unsigned short* qrow = Qg + (size_t)(q0 + lm) * DHEAD;
#pragma unroll
        for (int c = 0; c < 8; ++c) {
            qf[c].q[0] = *(const uint4*)(qrow + c * 32 + 16 * half);
            qf[c].q[1] = *(const uint4*)(qrow + c * 32 + 16 * half + 8);
#pragma unroll
            for (int i = 0; i < 16; ++i)
                qf[c].h[i] = f32_to_bf16(bf16_to_f32(qf[c].h[i]) * 0.0625f);
        }
    }

    v8f o[16];
#pragma unroll
    for (int t = 0; t < 16; ++t)
#pragma unroll
        for (int i = 0; i < 8; ++i) o[t][i] = 0.0f;
    float mrow[8], lrow[8];
#pragma unroll
    for (int r = 0; r < 8; ++r) { mrow[r] = -3.0e38f; lrow[r] = 0.0f; }

    // Prologue: stage block 0 into buffer 0.
    stage_v(Vg, lds_v[0]);
    stage_k(Kg, lds_k[0]);

    for (int kb = 0; kb < SEQ; kb += 32) {
        const int buf = (kb >> 5) & 1;
        __syncthreads();   // all waves done reading buf^1 (prev block compute)

        // Issue next block into the other buffer, then wait for the current
        // block: async/TDM ops complete in order per wave, so waiting down to
        // the just-issued count guarantees the previous block has landed.
        if (kb + 32 < SEQ) {
            stage_v(Vg + (size_t)(kb + 32) * DHEAD, lds_v[buf ^ 1]);
            stage_k(Kg + (size_t)(kb + 32) * DHEAD, lds_k[buf ^ 1]);
#if HAVE_TDM
            WAIT_ASYNC(8);
            if (wave == 0) wait_tensor();
#else
            WAIT_ASYNC(16);
#endif
        } else {
            WAIT_ASYNC(0);
#if HAVE_TDM
            if (wave == 0) wait_tensor0();
#endif
        }
        __syncthreads();   // block kb resident in lds_k[buf] / lds_v[buf]

        // ---- Transpose V block: lds_v[buf] (row-major) -> lds_vt ----
#pragma unroll
        for (int j = 0; j < 8; ++j) {
            const int e0 = (tid + j * 128) * 8;
            U4H u; u.q = *(const uint4*)&lds_v[buf][e0];
            const int row = e0 >> 8;      // key within block
            const int col = e0 & 255;     // head-dim
#pragma unroll
            for (int t = 0; t < 8; ++t)
                lds_vt[(col + t) * VT_STRIDE + row] = u.h[t];
        }
        __syncthreads();

        // ---- Scores: S = Q * K^T for 32 keys (two 16x16 tiles) ----
        v8f s0, s1;
#pragma unroll
        for (int i = 0; i < 8; ++i) { s0[i] = 0.0f; s1[i] = 0.0f; }
#pragma unroll
        for (int c = 0; c < 8; ++c) {
            const unsigned short* kr0 = &lds_k[buf][(size_t)lm * DHEAD + c * 32 + 16 * half];
            const unsigned short* kr1 = &lds_k[buf][(size_t)(16 + lm) * DHEAD + c * 32 + 16 * half];
            FragBF kf0, kf1;
            kf0.q[0] = *(const uint4*)(kr0);  kf0.q[1] = *(const uint4*)(kr0 + 8);
            kf1.q[0] = *(const uint4*)(kr1);  kf1.q[1] = *(const uint4*)(kr1 + 8);
            s0 = wmma_bf16(qf[c].v, kf0.v, s0);
            s1 = wmma_bf16(qf[c].v, kf1.v, s1);
        }

        // ---- Streaming softmax (row stats live in 16-lane half groups) ----
        float alpha[8];
#pragma unroll
        for (int r = 0; r < 8; ++r) {
            float bm = fmaxf(s0[r], s1[r]);
            bm = fmaxf(bm, __shfl_xor(bm, 1, 16));
            bm = fmaxf(bm, __shfl_xor(bm, 2, 16));
            bm = fmaxf(bm, __shfl_xor(bm, 4, 16));
            bm = fmaxf(bm, __shfl_xor(bm, 8, 16));
            const float mn = fmaxf(mrow[r], bm);
            const float a  = __expf(mrow[r] - mn);
            const float p0 = __expf(s0[r] - mn);
            const float p1 = __expf(s1[r] - mn);
            float rs = p0 + p1;
            rs += __shfl_xor(rs, 1, 16);
            rs += __shfl_xor(rs, 2, 16);
            rs += __shfl_xor(rs, 4, 16);
            rs += __shfl_xor(rs, 8, 16);
            lrow[r] = lrow[r] * a + rs;
            mrow[r] = mn;
            alpha[r] = a;
            s0[r] = p0;
            s1[r] = p1;
        }
#pragma unroll
        for (int t = 0; t < 16; ++t)
#pragma unroll
            for (int r = 0; r < 8; ++r) o[t][r] *= alpha[r];

        // ---- P: C-layout -> LDS -> A-layout (same-wave, DS ops in order) ----
        unsigned short* pb = &lds_p[wave * 16 * 32];
#pragma unroll
        for (int r = 0; r < 8; ++r) {
            const int row = r + 8 * half;
            pb[row * 32 + lm]      = f32_to_bf16(s0[r]);
            pb[row * 32 + 16 + lm] = f32_to_bf16(s1[r]);
        }
        FragBF pf;
        pf.q[0] = *(const uint4*)(pb + lm * 32 + 16 * half);
        pf.q[1] = *(const uint4*)(pb + lm * 32 + 16 * half + 8);

        // ---- O += P * V  (B fragments from transposed-V LDS) ----
#pragma unroll
        for (int t = 0; t < 16; ++t) {
            const unsigned short* vr = &lds_vt[(size_t)(t * 16 + lm) * VT_STRIDE + 16 * half];
            FragBF vf;
            vf.q[0] = *(const uint4*)(vr);
            vf.q[1] = *(const uint4*)(vr + 8);
            o[t] = wmma_bf16(pf.v, vf.v, o[t]);
        }
    }

    // ---- Finalize: O /= l, write attn_concat[b][s][h*256+d] as bf16 ----
    float inv[8];
#pragma unroll
    for (int r = 0; r < 8; ++r) inv[r] = (lrow[r] > 0.0f) ? 1.0f / lrow[r] : 0.0f;
#pragma unroll
    for (int t = 0; t < 16; ++t) {
        const int d = t * 16 + lm;
#pragma unroll
        for (int r = 0; r < 8; ++r) {
            const int row = r + 8 * half;
            const int s = q0 + row;
            size_t addr = ((size_t)b * SEQ + s) * HD + (size_t)h * DHEAD + d;
            obuf[addr] = f32_to_bf16(o[t][r] * inv[r]);
        }
    }
}

// ---------------------------------------------------------------------------
// Kernel 5: output projection + residual (software-pipelined K loop).
// ---------------------------------------------------------------------------
__global__ __launch_bounds__(128) void gemm_out_kernel(
        const unsigned short* __restrict__ Ob,
        const unsigned short* __restrict__ Wb,
        const float* __restrict__ bias,
        const float* __restrict__ x,
        float* __restrict__ out) {
    const int tid  = threadIdx.x;
    const int wave = tid >> 5;
    const int lane = tid & 31;
    const int lm   = lane & 15;
    const int half = lane >> 4;

    const int m0 = blockIdx.y * 64 + wave * 16;
    const int n0 = blockIdx.x * 64;

    v8f acc[4];
#pragma unroll
    for (int j = 0; j < 4; ++j)
#pragma unroll
        for (int i = 0; i < 8; ++i) acc[j][i] = 0.0f;

    const unsigned short* arow = Ob + (size_t)(m0 + lm) * HD;

    FragBF a[2], bmf[2][4];
    auto loadFrags = [&](int p, int kc) {
        a[p].q[0] = *(const uint4*)(arow + kc + 16 * half);
        a[p].q[1] = *(const uint4*)(arow + kc + 16 * half + 8);
#pragma unroll
        for (int j = 0; j < 4; ++j) {
            const unsigned short* wr =
                Wb + (size_t)(n0 + j * 16 + lm) * HD + kc + 16 * half;
            bmf[p][j].q[0] = *(const uint4*)(wr);
            bmf[p][j].q[1] = *(const uint4*)(wr + 8);
        }
    };

    loadFrags(0, 0);
    int p = 0;
    for (int kc = 0; kc < HD; kc += 32, p ^= 1) {
        if (kc + 32 < HD) loadFrags(p ^ 1, kc + 32);
#pragma unroll
        for (int j = 0; j < 4; ++j)
            acc[j] = wmma_bf16(a[p].v, bmf[p][j].v, acc[j]);
    }

#pragma unroll
    for (int j = 0; j < 4; ++j) {
        const int n = n0 + j * 16 + lm;
        const float bv = bias[n];
#pragma unroll
        for (int r = 0; r < 8; ++r) {
            const int row = m0 + r + 8 * half;
            const size_t addr = (size_t)row * DHEAD + n;
            out[addr] = acc[j][r] + bv + x[addr];
        }
    }
}

// ---------------------------------------------------------------------------
// Host launcher
// ---------------------------------------------------------------------------
extern "C" void kernel_launch(void* const* d_in, const int* in_sizes, int n_in,
                              void* d_out, int out_size, void* d_ws, size_t ws_size,
                              hipStream_t stream) {
    const float* x      = (const float*)d_in[0];   // [4,2048,256]
    const float* w_qkv  = (const float*)d_in[1];   // [6144,256]
    const float* b_qkv  = (const float*)d_in[2];   // [6144]
    const float* w_out  = (const float*)d_in[3];   // [256,2048]
    const float* b_out  = (const float*)d_in[4];   // [256]
    float* out = (float*)d_out;

    unsigned short* ws = (unsigned short*)d_ws;
    unsigned short* Xb    = ws;                                        // 8192*256
    unsigned short* Wqkvb = Xb    + (size_t)MROWS * DHEAD;             // 6144*256
    unsigned short* Woutb = Wqkvb + (size_t)NQKV * DHEAD;              // 256*2048
    unsigned short* QKVb  = Woutb + (size_t)DHEAD * HD;                // 3*B*H*S*D
    unsigned short* Ob    = QKVb  + (size_t)3 * BATCH * NHEAD * SEQ * DHEAD;

    cvt_bf16_kernel<<<2048, 256, 0, stream>>>(x,     Xb,    MROWS * DHEAD);
    cvt_bf16_kernel<<<2048, 256, 0, stream>>>(w_qkv, Wqkvb, NQKV * DHEAD);
    cvt_bf16_kernel<<<1024, 256, 0, stream>>>(w_out, Woutb, DHEAD * HD);

    gemm_qkv_kernel<<<dim3(NQKV / 64, MROWS / 64), 128, 0, stream>>>(
        Xb, Wqkvb, b_qkv, QKVb);

    rope_kernel<<<8192, 256, 0, stream>>>(QKVb);

    flash_attn_kernel<<<dim3(SEQ / 64, NHEAD, BATCH), 128, 0, stream>>>(QKVb, Ob);

    gemm_out_kernel<<<dim3(DHEAD / 64, MROWS / 64), 128, 0, stream>>>(
        Ob, Woutb, b_out, x, out);
}
